// MultiHeadAttention_86921548136611
// MI455X (gfx1250) — compile-verified, matrix operand order checked by reading, not run on hardware
//
#include <hip/hip_runtime.h>

// ---------------------------------------------------------------------------
// Types for CDNA5 WMMA (wave32): v_wmma_f32_16x16x32_bf16
// ---------------------------------------------------------------------------
typedef __attribute__((ext_vector_type(16))) __bf16         v16bf;
typedef __attribute__((ext_vector_type(8)))  float          v8f;
typedef __attribute__((ext_vector_type(8)))  unsigned short v8u;
typedef __attribute__((ext_vector_type(16))) unsigned short v16u;
typedef __attribute__((ext_vector_type(4)))  unsigned short v4u;
typedef int v4i_gcc __attribute__((vector_size(16)));   // matches builtin param

#define AS1 __attribute__((address_space(1)))
#define AS3 __attribute__((address_space(3)))

#define D_MODEL 1024
#define T_SEQ   2048
#define NHEAD   16
#define DK      64
#define BATCH   2
#define M_TOT   (BATCH * T_SEQ)   // 4096 rows in flattened (B,T)

__device__ __forceinline__ unsigned short f2bf(float f) {
    unsigned int u = __float_as_uint(f);
    u += 0x7FFFu + ((u >> 16) & 1u);      // round-to-nearest-even
    return (unsigned short)(u >> 16);
}

// Load one 16-bit WMMA operand fragment: lane holds two contiguous 8-element
// (16-byte) chunks at p and p+delta (ISA 16-bit A/B per-lane layout with
// p = base + (row_or_col)*stride + (lane>>4)*8, delta = 16).
__device__ __forceinline__ v16bf load_frag(const unsigned short* p, int delta) {
    v8u lo = *(const v8u*)(p);
    v8u hi = *(const v8u*)(p + delta);
    v16u f = __builtin_shufflevector(lo, hi, 0,1,2,3,4,5,6,7,8,9,10,11,12,13,14,15);
    return __builtin_bit_cast(v16bf, f);
}

__device__ __forceinline__ v8f wmma_bf16(v16bf a, v16bf b, v8f c) {
    return __builtin_amdgcn_wmma_f32_16x16x32_bf16(
        false, a, false, b, (short)0, c, false, false);
}

// --- CDNA5 async global->LDS DMA (ASYNCcnt), with synchronous fallback ------
__device__ __forceinline__ void async_copy_b128(const unsigned short* g,
                                                unsigned short* l) {
#if __has_builtin(__builtin_amdgcn_global_load_async_to_lds_b128)
    __builtin_amdgcn_global_load_async_to_lds_b128(
        (AS1 v4i_gcc*)g, (AS3 v4i_gcc*)l, 0, 0);
#else
    *(v8u*)l = *(const v8u*)g;
#endif
}

__device__ __forceinline__ void wait_async() {
#if __has_builtin(__builtin_amdgcn_global_load_async_to_lds_b128)
#if __has_builtin(__builtin_amdgcn_s_wait_asynccnt)
    __builtin_amdgcn_s_wait_asynccnt(0);
#else
    asm volatile("s_wait_asynccnt 0" ::: "memory");
#endif
#endif
}

__device__ __forceinline__ float rmax16(float v) {
    #pragma unroll
    for (int o = 1; o < 16; o <<= 1) v = fmaxf(v, __shfl_xor(v, o, 16));
    return v;
}
__device__ __forceinline__ float rsum16(float v) {
    #pragma unroll
    for (int o = 1; o < 16; o <<= 1) v += __shfl_xor(v, o, 16);
    return v;
}

// ---------------------------------------------------------------------------
// Kernel 1: f32 -> bf16 elementwise (vectorized x4)
// ---------------------------------------------------------------------------
__global__ void k_f32_to_bf16(const float* __restrict__ src,
                              unsigned short* __restrict__ dst, int n) {
    int i = (blockIdx.x * blockDim.x + threadIdx.x) * 4;
    if (i + 3 < n) {
        float4 v = *(const float4*)(src + i);
        v4u o;
        o[0] = f2bf(v.x); o[1] = f2bf(v.y); o[2] = f2bf(v.z); o[3] = f2bf(v.w);
        *(v4u*)(dst + i) = o;
    }
}

// ---------------------------------------------------------------------------
// Kernel 2: transpose + convert: W[K][N] f32 -> WT[N][K] bf16
// ---------------------------------------------------------------------------
__global__ void k_transpose_bf16(const float* __restrict__ W,
                                 unsigned short* __restrict__ WT,
                                 int K, int N) {
    int tx = threadIdx.x & 31;       // along n (coalesced read)
    int ty = threadIdx.x >> 5;       // along k
    int n = blockIdx.x * 32 + tx;
    int k = blockIdx.y * 8 + ty;
    if (n < N && k < K) WT[n * K + k] = f2bf(W[k * N + n]);
}

// ---------------------------------------------------------------------------
// GEMM core: one wave computes a 32x64 tile; block = 8 waves = 256x64.
// B tile (64 cols x 32 k) staged via async global->LDS double buffer.
// A fragments software-pipelined one k-step ahead in registers.
// acc[rr][j]: rr = row half (m0+rr*16), j = 16-col group.
// ---------------------------------------------------------------------------
struct GemmAcc { v8f a[2][4]; };

__device__ __forceinline__ void gemm_core_32x64(
        const unsigned short* __restrict__ A, int K,
        const unsigned short* __restrict__ bsrc,   // per-thread B stage src
        unsigned short* Bs0, unsigned short* Bs1,  // LDS buffers
        int stageOff,                              // per-thread LDS stage offset
        int m0, int l15, int halo, GemmAcc& acc) {
    #pragma unroll
    for (int rr = 0; rr < 2; ++rr)
        #pragma unroll
        for (int j = 0; j < 4; ++j) acc.a[rr][j] = v8f{0,0,0,0,0,0,0,0};

    async_copy_b128(bsrc, Bs0 + stageOff);

    const unsigned short* pa0 = A + (m0 +      l15) * K + halo;
    const unsigned short* pa1 = A + (m0 + 16 + l15) * K + halo;
    v16bf a0 = load_frag(pa0, 16);
    v16bf a1 = load_frag(pa1, 16);

    for (int kk = 0; kk < K; kk += 32) {
        unsigned short* cur = ((kk >> 5) & 1) ? Bs1 : Bs0;
        unsigned short* nxt = ((kk >> 5) & 1) ? Bs0 : Bs1;
        wait_async();
        __syncthreads();
        if (kk + 32 < K)
            async_copy_b128(bsrc + kk + 32, nxt + stageOff);

        // prefetch next A fragments (covered by this iteration's compute)
        int kn = (kk + 32 < K) ? kk + 32 : kk;
        v16bf na0 = load_frag(pa0 + kn, 16);
        v16bf na1 = load_frag(pa1 + kn, 16);

        // batch all B fragment LDS reads, then the 8 WMMAs
        v16bf b0 = load_frag(cur + (0 * 16 + l15) * 32 + halo, 16);
        v16bf b1 = load_frag(cur + (1 * 16 + l15) * 32 + halo, 16);
        v16bf b2 = load_frag(cur + (2 * 16 + l15) * 32 + halo, 16);
        v16bf b3 = load_frag(cur + (3 * 16 + l15) * 32 + halo, 16);

        acc.a[0][0] = wmma_bf16(a0, b0, acc.a[0][0]);
        acc.a[0][1] = wmma_bf16(a0, b1, acc.a[0][1]);
        acc.a[0][2] = wmma_bf16(a0, b2, acc.a[0][2]);
        acc.a[0][3] = wmma_bf16(a0, b3, acc.a[0][3]);
        acc.a[1][0] = wmma_bf16(a1, b0, acc.a[1][0]);
        acc.a[1][1] = wmma_bf16(a1, b1, acc.a[1][1]);
        acc.a[1][2] = wmma_bf16(a1, b2, acc.a[1][2]);
        acc.a[1][3] = wmma_bf16(a1, b3, acc.a[1][3]);

        a0 = na0;
        a1 = na1;
    }
}

// ---------------------------------------------------------------------------
// Kernel 3: QKV projection. Grid: 16 Mblocks(256) x 48 Nblocks(64).
// Epilogue scatters bf16: Q,K -> [B,H,T,64]; V -> transposed [B,H,64,T].
// ---------------------------------------------------------------------------
__global__ __launch_bounds__(256)
void k_qkv_gemm(const unsigned short* __restrict__ xb,
                const unsigned short* __restrict__ wT,
                const float* __restrict__ bqkv,
                unsigned short* __restrict__ qb,
                unsigned short* __restrict__ kb,
                unsigned short* __restrict__ vtb) {
    const int K = D_MODEL;
    __shared__ unsigned short Bs[2][64 * 32];

    int wv   = threadIdx.x >> 5;
    int lane = threadIdx.x & 31;
    int l15  = lane & 15;
    int halo = (lane >> 4) << 3;

    int bm = blockIdx.x & 15;        // 16 M-blocks of 256
    int bn = blockIdx.x >> 4;        // 48 N-blocks of 64
    int m0 = bm * 256 + wv * 32;
    int n0 = bn * 64;

    int trow = threadIdx.x >> 2;            // 0..63  (column n within tile)
    int tcol = (threadIdx.x & 3) << 3;      // 0,8,16,24 (k within 32-chunk)
    const unsigned short* bsrc = wT + (n0 + trow) * K + tcol;
    int stageOff = trow * 32 + tcol;

    GemmAcc acc;
    gemm_core_32x64(xb, K, bsrc, Bs[0], Bs[1], stageOff, m0, l15, halo, acc);

    #pragma unroll
    for (int j = 0; j < 4; ++j) {
        int col  = n0 + j * 16 + l15;        // 0..3071
        float bias = bqkv[col];
        int sel = col >> 10;                 // 0=Q 1=K 2=V
        int rem = col & 1023;
        int h = rem >> 6;
        int d = rem & 63;
        #pragma unroll
        for (int rr = 0; rr < 2; ++rr) {
            #pragma unroll
            for (int r = 0; r < 8; ++r) {
                int m = m0 + rr * 16 + r + halo;     // flattened (b,t)
                int bidx = m >> 11;
                int t = m & (T_SEQ - 1);
                unsigned short bv = f2bf(acc.a[rr][j][r] + bias);
                int bh = bidx * NHEAD + h;
                if (sel == 0)      qb[(bh * T_SEQ + t) * DK + d] = bv;
                else if (sel == 1) kb[(bh * T_SEQ + t) * DK + d] = bv;
                else               vtb[(bh * DK + d) * T_SEQ + t] = bv;
            }
        }
    }
}

// ---------------------------------------------------------------------------
// Kernel 4: flash attention. All 8 waves of a block share one (b,h); each wave
// owns a 16-query tile. K tile (32x64) and V^T tile (64x32) staged via async
// global->LDS, double-buffered, shared by all 8 waves.
// ---------------------------------------------------------------------------
__global__ __launch_bounds__(256)
void k_attn(const unsigned short* __restrict__ qb,
            const unsigned short* __restrict__ kb,
            const unsigned short* __restrict__ vtb,
            unsigned short* __restrict__ attnb) {
    __shared__ unsigned short Ks[2][32 * 64];
    __shared__ unsigned short Vs[2][64 * 32];
    __shared__ unsigned short Ps[8][16 * 32];

    int wv   = threadIdx.x >> 5;
    int lane = threadIdx.x & 31;
    int l15  = lane & 15;
    int halo = (lane >> 4) << 3;

    int bh = blockIdx.x >> 4;                    // (b*16+h), uniform per block
    int qt = ((blockIdx.x & 15) << 3) + wv;      // query tile in [0,128)
    int m0 = qt * 16;

    const unsigned short* qp = qb  + (size_t)bh * T_SEQ * DK;
    const unsigned short* kp = kb  + (size_t)bh * T_SEQ * DK;
    const unsigned short* vp = vtb + (size_t)bh * DK * T_SEQ;

    int ktt = threadIdx.x >> 3;                  // 0..31 key row
    int kdc = (threadIdx.x & 7) << 3;            // 0..56 d chunk
    int vdd = threadIdx.x >> 2;                  // 0..63 d row
    int vtc = (threadIdx.x & 3) << 3;            // 0..24 t chunk

    v16bf qa0 = load_frag(qp + (m0 + l15) * DK + 0  + halo, 16);
    v16bf qa1 = load_frag(qp + (m0 + l15) * DK + 32 + halo, 16);

    float mr[8], lr[8];
    v8f o[4];
    #pragma unroll
    for (int r = 0; r < 8; ++r) { mr[r] = -3.402823466e38f; lr[r] = 0.f; }
    #pragma unroll
    for (int j = 0; j < 4; ++j) o[j] = v8f{0,0,0,0,0,0,0,0};

    async_copy_b128(kp + ktt * DK + kdc,      &Ks[0][ktt * 64 + kdc]);
    async_copy_b128(vp + vdd * T_SEQ + vtc,   &Vs[0][vdd * 32 + vtc]);

    for (int blk = 0; blk < T_SEQ / 32; ++blk) {
        int t0  = blk * 32;
        int buf = blk & 1;

        wait_async();
        __syncthreads();
        if (blk + 1 < T_SEQ / 32) {
            int tn = t0 + 32;
            async_copy_b128(kp + (tn + ktt) * DK + kdc,    &Ks[buf ^ 1][ktt * 64 + kdc]);
            async_copy_b128(vp + vdd * T_SEQ + tn + vtc,   &Vs[buf ^ 1][vdd * 32 + vtc]);
        }

        v8f s0 = v8f{0,0,0,0,0,0,0,0};
        v8f s1 = v8f{0,0,0,0,0,0,0,0};

        v16bf b00 = load_frag(&Ks[buf][(     l15) * 64 + 0  + halo], 16);
        v16bf b01 = load_frag(&Ks[buf][(     l15) * 64 + 32 + halo], 16);
        v16bf b10 = load_frag(&Ks[buf][(16 + l15) * 64 + 0  + halo], 16);
        v16bf b11 = load_frag(&Ks[buf][(16 + l15) * 64 + 32 + halo], 16);
        s0 = wmma_bf16(qa0, b00, s0);
        s0 = wmma_bf16(qa1, b01, s0);
        s1 = wmma_bf16(qa0, b10, s1);
        s1 = wmma_bf16(qa1, b11, s1);

        // --- online softmax (rows map to (vgpr r, lane-half)) ---
        float newm[8];
        #pragma unroll
        for (int r = 0; r < 8; ++r) {
            float a0 = s0[r] * 0.125f;      // 1/sqrt(64)
            float a1 = s1[r] * 0.125f;
            s0[r] = a0; s1[r] = a1;
            newm[r] = fmaxf(mr[r], rmax16(fmaxf(a0, a1)));
        }
        #pragma unroll
        for (int r = 0; r < 8; ++r) {
            float nm = newm[r];
            float p0 = __expf(s0[r] - nm);
            float p1 = __expf(s1[r] - nm);
            float rs = rsum16(p0 + p1);
            float sc = __expf(mr[r] - nm);
            lr[r] = lr[r] * sc + rs;
            mr[r] = nm;
            #pragma unroll
            for (int j = 0; j < 4; ++j) o[j][r] *= sc;
            int row = r + halo;
            Ps[wv][row * 32 +      l15] = f2bf(p0);
            Ps[wv][row * 32 + 16 + l15] = f2bf(p1);
        }

        // restage P as A fragment (16x32), batch V fragment reads, then PV
        v16bf pa  = load_frag(&Ps[wv][l15 * 32 + halo], 16);
        v16bf vb0 = load_frag(&Vs[buf][(0 * 16 + l15) * 32 + halo], 16);
        v16bf vb1 = load_frag(&Vs[buf][(1 * 16 + l15) * 32 + halo], 16);
        v16bf vb2 = load_frag(&Vs[buf][(2 * 16 + l15) * 32 + halo], 16);
        v16bf vb3 = load_frag(&Vs[buf][(3 * 16 + l15) * 32 + halo], 16);
        o[0] = wmma_bf16(pa, vb0, o[0]);
        o[1] = wmma_bf16(pa, vb1, o[1]);
        o[2] = wmma_bf16(pa, vb2, o[2]);
        o[3] = wmma_bf16(pa, vb3, o[3]);
    }

    // normalize + store bf16 to [B*T, 1024] with head-interleaved columns
    int h = bh & (NHEAD - 1);
    int b = bh >> 4;
    #pragma unroll
    for (int j = 0; j < 4; ++j) {
        #pragma unroll
        for (int r = 0; r < 8; ++r) {
            int m = m0 + r + halo;
            float v = o[j][r] / lr[r];
            attnb[(b * T_SEQ + m) * D_MODEL + h * DK + j * 16 + l15] = f2bf(v);
        }
    }
}

// ---------------------------------------------------------------------------
// Kernel 5: output projection. Grid: 16 Mblocks(256) x 16 Nblocks(64).
// ---------------------------------------------------------------------------
__global__ __launch_bounds__(256)
void k_out_gemm(const unsigned short* __restrict__ ab,
                const unsigned short* __restrict__ wT,
                const float* __restrict__ bout,
                float* __restrict__ out) {
    const int K = D_MODEL;
    __shared__ unsigned short Bs[2][64 * 32];

    int wv   = threadIdx.x >> 5;
    int lane = threadIdx.x & 31;
    int l15  = lane & 15;
    int halo = (lane >> 4) << 3;

    int bm = blockIdx.x & 15;        // 16 M-blocks of 256
    int bn = blockIdx.x >> 4;        // 16 N-blocks of 64
    int m0 = bm * 256 + wv * 32;
    int n0 = bn * 64;

    int trow = threadIdx.x >> 2;
    int tcol = (threadIdx.x & 3) << 3;
    const unsigned short* bsrc = wT + (n0 + trow) * K + tcol;
    int stageOff = trow * 32 + tcol;

    GemmAcc acc;
    gemm_core_32x64(ab, K, bsrc, Bs[0], Bs[1], stageOff, m0, l15, halo, acc);

    #pragma unroll
    for (int j = 0; j < 4; ++j) {
        int col = n0 + j * 16 + l15;
        float bias = bout[col];
        #pragma unroll
        for (int rr = 0; rr < 2; ++rr) {
            #pragma unroll
            for (int r = 0; r < 8; ++r) {
                int m = m0 + rr * 16 + r + halo;
                out[m * D_MODEL + col] = acc.a[rr][j][r] + bias;
            }
        }
    }
}

// ---------------------------------------------------------------------------
// Host-side launcher
// ---------------------------------------------------------------------------
extern "C" void kernel_launch(void* const* d_in, const int* in_sizes, int n_in,
                              void* d_out, int out_size, void* d_ws, size_t ws_size,
                              hipStream_t stream) {
    const float* x    = (const float*)d_in[0];  // [2,2048,1024]
    const float* Wqkv = (const float*)d_in[1];  // [1024,3072]
    const float* bqkv = (const float*)d_in[2];  // [3072]
    const float* Wout = (const float*)d_in[3];  // [1024,1024]
    const float* bout = (const float*)d_in[4];  // [1024]
    float* out = (float*)d_out;                 // [2,2048,1024]

    char* ws = (char*)d_ws;
    size_t off = 0;
    unsigned short* xb     = (unsigned short*)(ws + off); off += (size_t)M_TOT * D_MODEL * 2;
    unsigned short* wqkvT  = (unsigned short*)(ws + off); off += (size_t)3 * D_MODEL * D_MODEL * 2;
    unsigned short* woutT  = (unsigned short*)(ws + off); off += (size_t)D_MODEL * D_MODEL * 2;
    unsigned short* qb     = (unsigned short*)(ws + off); off += (size_t)M_TOT * D_MODEL * 2;
    unsigned short* kb     = (unsigned short*)(ws + off); off += (size_t)M_TOT * D_MODEL * 2;
    unsigned short* vtb    = (unsigned short*)(ws + off); off += (size_t)M_TOT * D_MODEL * 2;
    unsigned short* attnb  = (unsigned short*)(ws + off); off += (size_t)M_TOT * D_MODEL * 2;

    {
        int n = M_TOT * D_MODEL;
        k_f32_to_bf16<<<n / (256 * 4), 256, 0, stream>>>(x, xb, n);
    }
    k_transpose_bf16<<<dim3(3 * D_MODEL / 32, D_MODEL / 8), 256, 0, stream>>>(Wqkv, wqkvT, D_MODEL, 3 * D_MODEL);
    k_transpose_bf16<<<dim3(D_MODEL / 32, D_MODEL / 8), 256, 0, stream>>>(Wout, woutT, D_MODEL, D_MODEL);

    // 16 Mblocks * 48 Nblocks
    k_qkv_gemm<<<768, 256, 0, stream>>>(xb, wqkvT, bqkv, qb, kb, vtb);

    // 2*16 heads * 16 blocks (8 query tiles each) = 512 blocks
    k_attn<<<512, 256, 0, stream>>>(qb, kb, vtb, attnb);

    // 16 Mblocks * 16 Nblocks
    k_out_gemm<<<256, 256, 0, stream>>>(attnb, woutT, bout, out);
}
